// TransformerCAModule_29789893165200
// MI455X (gfx1250) — matrix operand order repaired and checked by reading, not compiled, not run on hardware
//
#include <hip/hip_runtime.h>
#include <hip/hip_bf16.h>

typedef _Float16 f16_t;
typedef __attribute__((ext_vector_type(16))) _Float16 v16h;
typedef __attribute__((ext_vector_type(8)))  _Float16 v8h;
typedef __attribute__((ext_vector_type(8)))  float    v8f;

#define DMODEL 128
#define NHEAD  4
#define HDIM   32
#define DFF    256
#define NLAYER 2
#define TOK    9
#define WPB    7          // windows per block (63 tokens -> 64 padded)
#define MT     64         // padded token rows per block
#define MTILES (MT / 16)  // 4 M-tiles
#define HS     136        // hbuf row stride (halves), pad vs bank conflicts
#define QS     392        // qkv/mid row stride (halves)
#define TS     132        // residual row stride (floats)
#define HPAD   130        // padded image dim
#define EPAD_ELEMS (2*HPAD*HPAD*DMODEL)   // 4,326,400 halves
#define PACK_PER_LAYER 131072             // halves

#define ASYNC_STAGE 1

// async copy 16B global -> LDS via the CDNA5 async engine (ASYNCcnt)
__device__ __forceinline__ void async_load_b128(const void* gp, void* lp) {
  unsigned int loff = (unsigned int)(uintptr_t)lp;        // LDS byte address
  unsigned long long ga = (unsigned long long)(uintptr_t)gp;
  asm volatile("global_load_async_to_lds_b128 %0, %1, off"
               :: "v"(loff), "v"(ga) : "memory");
}

__device__ __forceinline__ void wait_async0() {
#if __has_builtin(__builtin_amdgcn_s_wait_asynccnt)
  __builtin_amdgcn_s_wait_asynccnt(0);
#else
  asm volatile("s_wait_asynccnt 0" ::: "memory");
#endif
}

// ---------------- prep: embedding into zero-padded f16 map ----------------
__global__ void embed_pad_kernel(const float* __restrict__ x,
                                 const float* __restrict__ emb_w,
                                 const float* __restrict__ emb_b,
                                 f16_t* __restrict__ epad) {
  int idx = blockIdx.x * 256 + threadIdx.x;
  if (idx >= EPAD_ELEMS) return;
  int d  = idx & 127;
  int t  = idx >> 7;
  int xx = t % HPAD; t /= HPAD;
  int yy = t % HPAD;
  int b  = t / HPAD;
  float v = 0.f;
  if (yy >= 1 && yy <= 128 && xx >= 1 && xx <= 128) {
    int y = yy - 1, xq = xx - 1;
    float acc = emb_b[d];
#pragma unroll
    for (int c = 0; c < 16; ++c)
      acc += x[((b * 16 + c) * 128 + y) * 128 + xq] * emb_w[d * 16 + c];
    v = acc;
  }
  epad[idx] = (f16_t)v;
}

// ------- prep: pack W^T into per-lane WMMA B-fragment layout (f16) --------
// B[k][n] = W[n][k].  dst[((kt*NT+nt)*32+lane)*16 + j],
// k = kt*32 + (lane>>4)*8 + (j&7) + (j>>3)*16,  n = nt*16 + (lane&15).
__global__ void pack_b_kernel(const float* __restrict__ W, int Ksrc,
                              int KT, int NT, f16_t* __restrict__ dst) {
  int idx = blockIdx.x * 256 + threadIdx.x;
  int total = KT * NT * 512;
  if (idx >= total) return;
  int j    = idx & 15;
  int lane = (idx >> 4) & 31;
  int tile = idx >> 9;
  int nt = tile % NT, kt = tile / NT;
  int k   = kt * 32 + ((lane >> 4) << 3) + (j & 7) + ((j >> 3) << 4);
  int col = nt * 16 + (lane & 15);
  dst[idx] = (f16_t)W[col * Ksrc + k];
}

// ---------------- fragment loads ----------------
__device__ __forceinline__ v16h load_a_frag(const f16_t* buf, int stride,
                                            int mt, int kt, int lane) {
  int row = mt * 16 + (lane & 15);
  int k0  = kt * 32 + ((lane >> 4) << 3);
  const f16_t* base = buf + row * stride + k0;
  v8h lo = *(const v8h*)(base);
  v8h hi = *(const v8h*)(base + 16);
  return __builtin_shufflevector(lo, hi, 0,1,2,3,4,5,6,7,8,9,10,11,12,13,14,15);
}

__device__ __forceinline__ v16h load_b_frag(const f16_t* pack, int NT,
                                            int kt, int nt, int lane) {
  return *(const v16h*)(pack + ((((kt * NT + nt) * 32) + lane) << 4));
}

// ---------------- generic WMMA GEMM stage ----------------
// MODE 0: dstH[row][col] = f16(acc + bias)          (0 for pad rows)
// MODE 1: Tb[row][col]  += acc + bias               (skip pad rows)
// MODE 2: dstH[row][col] = f16(gelu(acc + bias))    (0 for pad rows)
template <int MODE>
__device__ __forceinline__ void gemm_stage(const f16_t* __restrict__ A, int As,
                                           const f16_t* __restrict__ Bp,
                                           int KT, int NT,
                                           const float* __restrict__ bias,
                                           f16_t* __restrict__ dstH, int dstS,
                                           float* __restrict__ Tb,
                                           int ntok, int lane, int wave) {
  int total = MTILES * NT;
  for (int t = wave; t < total; t += 8) {
    int mt = t / NT, nt = t % NT;
    v8f c = {};
    for (int kt = 0; kt < KT; ++kt) {
      if (kt + 1 < KT)
        __builtin_prefetch((const void*)(Bp + (((((kt + 1) * NT + nt) * 32) + lane) << 4)), 0, 1);
      v16h a = load_a_frag(A, As, mt, kt, lane);
      v16h b = load_b_frag(Bp, NT, kt, nt, lane);
      c = __builtin_amdgcn_wmma_f32_16x16x32_f16(false, a, false, b,
                                                 (short)0, c, false, false);
    }
    int col   = nt * 16 + (lane & 15);
    float bv  = bias[col];
    int rbase = mt * 16 + ((lane >> 4) << 3);
#pragma unroll
    for (int r = 0; r < 8; ++r) {
      int row = rbase + r;
      float v = c[r] + bv;
      if (MODE == 0) {
        dstH[row * dstS + col] = (f16_t)((row < ntok) ? v : 0.f);
      } else if (MODE == 1) {
        if (row < ntok) Tb[row * TS + col] += v;
      } else {
        float g = 0.5f * v * (1.f + erff(v * 0.70710678118f));
        dstH[row * dstS + col] = (f16_t)((row < ntok) ? g : 0.f);
      }
    }
  }
}

// ---------------- LayerNorm: T(f32) -> h(f16), 8 lanes per token ----------
__device__ __forceinline__ void ln_stage(const float* __restrict__ Tb,
                                         f16_t* __restrict__ hb,
                                         const float* __restrict__ g,
                                         const float* __restrict__ b,
                                         int ntok, int tid) {
  int sub = tid & 7;
  for (int token = tid >> 3; token < ntok; token += 32) {
    const float* tr = Tb + token * TS + sub * 16;
    float vals[16];
    float s = 0.f, ss = 0.f;
#pragma unroll
    for (int i = 0; i < 16; ++i) { float v = tr[i]; vals[i] = v; s += v; ss += v * v; }
#pragma unroll
    for (int m = 1; m < 8; m <<= 1) { s += __shfl_xor(s, m, 32); ss += __shfl_xor(ss, m, 32); }
    float mean = s * (1.f / 128.f);
    float rstd = rsqrtf(ss * (1.f / 128.f) - mean * mean + 1e-5f);
    f16_t* hr = hb + token * HS + sub * 16;
#pragma unroll
    for (int i = 0; i < 16; ++i) {
      int col = sub * 16 + i;
      hr[i] = (f16_t)((vals[i] - mean) * rstd * g[col] + b[col]);
    }
  }
}

// ---------------- fused per-window transformer ----------------
__global__ __launch_bounds__(256)
void fused_win_transformer(const f16_t* __restrict__ epad,
                           const f16_t* __restrict__ packs,
                           const float* __restrict__ pos,
                           const float* __restrict__ bqkv,
                           const float* __restrict__ bo,
                           const float* __restrict__ ln1_g, const float* __restrict__ ln1_b,
                           const float* __restrict__ ln2_g, const float* __restrict__ ln2_b,
                           const float* __restrict__ b1, const float* __restrict__ b2,
                           const float* __restrict__ lnf_g, const float* __restrict__ lnf_b,
                           const float* __restrict__ head_w, const float* __restrict__ head_b,
                           float* __restrict__ out, int nwin) {
  __shared__ float Tbuf[MT * TS];   // residual stream, f32 (33 KB)
  __shared__ f16_t hbuf[MT * HS];   // LN out / attention out (17 KB)
  __shared__ f16_t qbuf[MT * QS];   // qkv / FFN mid / init staging (49 KB)

  const int tid  = threadIdx.x;
  const int lane = tid & 31;
  const int wave = tid >> 5;
  const int wbase = blockIdx.x * WPB;
  int wcount = nwin - wbase;
  if (wcount > WPB) wcount = WPB;
  if (wcount < 0) wcount = 0;
  const int ntok = wcount * TOK;

#if ASYNC_STAGE
  // stage window token rows (f16, 256B each) into LDS via the async engine
  for (int idx = tid; idx < ntok * 16; idx += 256) {
    int row = idx >> 4;
    int wid = wbase + row / 9;
    int tk  = row % 9;
    int b   = wid >> 14;
    int rem = wid & 16383;
    int y = rem >> 7, xq = rem & 127;
    const f16_t* gp = epad +
        ((b * HPAD + (y + tk / 3)) * HPAD + (xq + tk % 3)) * DMODEL + (idx & 15) * 8;
    async_load_b128((const void*)gp, (void*)(qbuf + idx * 8));
  }
  wait_async0();
  __syncthreads();
  for (int idx = tid; idx < MT * DMODEL; idx += 256) {
    int row = idx >> 7, col = idx & 127;
    float v = 0.f;
    if (row < ntok) {
      int tk = row % 9;
      v = (float)qbuf[row * DMODEL + col] + pos[tk * DMODEL + col];
    }
    Tbuf[row * TS + col] = v;
  }
#else
  for (int idx = tid; idx < MT * DMODEL; idx += 256) {
    int row = idx >> 7, col = idx & 127;
    float v = 0.f;
    if (row < ntok) {
      int wid = wbase + row / 9;
      int tk  = row % 9;
      int b   = wid >> 14;
      int rem = wid & 16383;
      int y = rem >> 7, xq = rem & 127;
      int ei = ((b * HPAD + (y + tk / 3)) * HPAD + (xq + tk % 3)) * DMODEL + col;
      v = (float)epad[ei] + pos[tk * DMODEL + col];
    }
    Tbuf[row * TS + col] = v;
  }
#endif
  for (int idx = tid; idx < MT * HS; idx += 256) hbuf[idx] = (f16_t)0.f;
  __syncthreads();

  for (int l = 0; l < NLAYER; ++l) {
    const f16_t* pk     = packs + l * PACK_PER_LAYER;
    const f16_t* pkqkv  = pk;            // 4*24*512 = 49152
    const f16_t* pkwo   = pk + 49152;    // 4*8*512  = 16384
    const f16_t* pkw1   = pk + 65536;    // 4*16*512 = 32768
    const f16_t* pkw2   = pk + 98304;    // 8*8*512  = 32768

    // h = LN1(T)
    ln_stage(Tbuf, hbuf, ln1_g + l * DMODEL, ln1_b + l * DMODEL, ntok, tid);
    __syncthreads();

    // qkv = h @ Wqkv^T + bqkv    (M x 384, K = 128)
    gemm_stage<0>(hbuf, HS, pkqkv, 4, 24, bqkv + l * 3 * DMODEL,
                  qbuf, QS, nullptr, ntok, lane, wave);
    __syncthreads();

    // attention: one thread per (window, head, query-row)  (252/256 active)
    if (tid < wcount * NHEAD * TOK) {
      int w = tid / (NHEAD * TOK);
      int rem = tid % (NHEAD * TOK);
      int h = rem / TOK, q = rem % TOK;
      int qrow = w * TOK + q;
      float qv[HDIM];
      const f16_t* qp = &qbuf[qrow * QS + h * HDIM];
#pragma unroll
      for (int d = 0; d < HDIM; ++d) qv[d] = (float)qp[d];
      float sc[TOK];
      float mx = -1e30f;
      for (int kk = 0; kk < TOK; ++kk) {
        const f16_t* kp = &qbuf[(w * TOK + kk) * QS + DMODEL + h * HDIM];
        float s = 0.f;
#pragma unroll
        for (int d = 0; d < HDIM; ++d) s += qv[d] * (float)kp[d];
        s *= 0.17677669529663687f;   // 1/sqrt(32)
        sc[kk] = s;
        mx = fmaxf(mx, s);
      }
      float se = 0.f;
#pragma unroll
      for (int kk = 0; kk < TOK; ++kk) { sc[kk] = __expf(sc[kk] - mx); se += sc[kk]; }
      float inv = 1.f / se;
      float ov[HDIM];
#pragma unroll
      for (int d = 0; d < HDIM; ++d) ov[d] = 0.f;
      for (int kk = 0; kk < TOK; ++kk) {
        const f16_t* vp = &qbuf[(w * TOK + kk) * QS + 2 * DMODEL + h * HDIM];
        float a = sc[kk] * inv;
#pragma unroll
        for (int d = 0; d < HDIM; ++d) ov[d] += a * (float)vp[d];
      }
      f16_t* op = &hbuf[qrow * HS + h * HDIM];
#pragma unroll
      for (int d = 0; d < HDIM; ++d) op[d] = (f16_t)ov[d];
    }
    __syncthreads();

    // T += o @ Wo^T + bo
    gemm_stage<1>(hbuf, HS, pkwo, 4, 8, bo + l * DMODEL,
                  nullptr, 0, Tbuf, ntok, lane, wave);
    __syncthreads();

    // h2 = LN2(T)
    ln_stage(Tbuf, hbuf, ln2_g + l * DMODEL, ln2_b + l * DMODEL, ntok, tid);
    __syncthreads();

    // mid = gelu(h2 @ W1^T + b1)   (M x 256, K = 128)
    gemm_stage<2>(hbuf, HS, pkw1, 4, 16, b1 + l * DFF,
                  qbuf, QS, nullptr, ntok, lane, wave);
    __syncthreads();

    // T += mid @ W2^T + b2         (M x 128, K = 256)
    gemm_stage<1>(qbuf, QS, pkw2, 8, 8, b2 + l * DMODEL,
                  nullptr, 0, Tbuf, ntok, lane, wave);
    __syncthreads();
  }

  // final: head on center token (one wave per window)
  if (wave < wcount) {
    int row = wave * TOK + 4;
    const float* tr = &Tbuf[row * TS];
    float s = 0.f, ss = 0.f;
#pragma unroll
    for (int i = 0; i < 4; ++i) { float v = tr[lane * 4 + i]; s += v; ss += v * v; }
#pragma unroll
    for (int m = 1; m < 32; m <<= 1) { s += __shfl_xor(s, m, 32); ss += __shfl_xor(ss, m, 32); }
    float mean = s * (1.f / 128.f);
    float rstd = rsqrtf(ss * (1.f / 128.f) - mean * mean + 1e-5f);
    float dot = 0.f;
#pragma unroll
    for (int i = 0; i < 4; ++i) {
      int col = lane * 4 + i;
      float nv = (tr[col] - mean) * rstd * lnf_g[col] + lnf_b[col];
      dot += nv * head_w[col];
    }
#pragma unroll
    for (int m = 1; m < 32; m <<= 1) dot += __shfl_xor(dot, m, 32);
    if (lane == 0) out[wbase + wave] = dot + head_b[0];
  }
}

// ---------------- host launcher ----------------
extern "C" void kernel_launch(void* const* d_in, const int* in_sizes, int n_in,
                              void* d_out, int out_size, void* d_ws, size_t ws_size,
                              hipStream_t stream) {
  (void)in_sizes; (void)n_in; (void)out_size; (void)ws_size;
  const float* x      = (const float*)d_in[0];
  const float* emb_w  = (const float*)d_in[1];
  const float* emb_b  = (const float*)d_in[2];
  const float* pos    = (const float*)d_in[3];
  const float* wqkv   = (const float*)d_in[4];
  const float* bqkv   = (const float*)d_in[5];
  const float* wo     = (const float*)d_in[6];
  const float* bo     = (const float*)d_in[7];
  const float* ln1_g  = (const float*)d_in[8];
  const float* ln1_b  = (const float*)d_in[9];
  const float* ln2_g  = (const float*)d_in[10];
  const float* ln2_b  = (const float*)d_in[11];
  const float* w1     = (const float*)d_in[12];
  const float* b1     = (const float*)d_in[13];
  const float* w2     = (const float*)d_in[14];
  const float* b2     = (const float*)d_in[15];
  const float* lnf_g  = (const float*)d_in[16];
  const float* lnf_b  = (const float*)d_in[17];
  const float* head_w = (const float*)d_in[18];
  const float* head_b = (const float*)d_in[19];

  f16_t* ws    = (f16_t*)d_ws;
  f16_t* epad  = ws;                       // EPAD_ELEMS halves
  f16_t* packs = ws + EPAD_ELEMS;          // 2 * PACK_PER_LAYER halves

  embed_pad_kernel<<<(EPAD_ELEMS + 255) / 256, 256, 0, stream>>>(x, emb_w, emb_b, epad);

  for (int l = 0; l < NLAYER; ++l) {
    f16_t* pk = packs + l * PACK_PER_LAYER;
    pack_b_kernel<<<(4 * 24 * 512 + 255) / 256, 256, 0, stream>>>(
        wqkv + l * 3 * DMODEL * DMODEL, DMODEL, 4, 24, pk);
    pack_b_kernel<<<(4 * 8 * 512 + 255) / 256, 256, 0, stream>>>(
        wo + l * DMODEL * DMODEL, DMODEL, 4, 8, pk + 49152);
    pack_b_kernel<<<(4 * 16 * 512 + 255) / 256, 256, 0, stream>>>(
        w1 + l * DFF * DMODEL, DMODEL, 4, 16, pk + 65536);
    pack_b_kernel<<<(8 * 8 * 512 + 255) / 256, 256, 0, stream>>>(
        w2 + l * DMODEL * DFF, DFF, 8, 8, pk + 98304);
  }

  int nwin = 2 * 128 * 128;
  fused_win_transformer<<<(nwin + WPB - 1) / WPB, 256, 0, stream>>>(
      epad, packs, pos, bqkv, bo, ln1_g, ln1_b, ln2_g, ln2_b,
      b1, b2, lnf_g, lnf_b, head_w, head_b, (float*)d_out, nwin);
}
